// L2GatedGraphConv_84859963834415
// MI455X (gfx1250) — compile-verified, matrix operand order checked by reading, not run on hardware
//
#include <hip/hip_runtime.h>

#define CCH 128
#define C3CH 384

typedef __attribute__((ext_vector_type(16))) __bf16 v16bf;
typedef __attribute__((ext_vector_type(8)))  __bf16 v8bf;
typedef __attribute__((ext_vector_type(4)))  __bf16 v4bf;
typedef __attribute__((ext_vector_type(8)))  float  v8f;

// Fragment geometry: K=128 -> 4 k-tiles of 32; N -> NT n-tiles of 16.
// B fragment store order: [kt][nt][lane][16 bf16]  (32B per lane => v16bf load)
#define FRAG_ELEMS 16

// ---------------------------------------------------------------------------
// WMMA helper: D = A(16x32 bf16) * B(32x16 bf16) + C(16x16 f32)
// ---------------------------------------------------------------------------
__device__ __forceinline__ v8f wmma_bf16(v16bf a, v16bf b, v8f c) {
  return __builtin_amdgcn_wmma_f32_16x16x32_bf16(
      /*neg_a=*/false, a, /*neg_b=*/false, b,
      /*c_mod=*/(short)0, c, /*reuse_a=*/false, /*reuse_b=*/false);
}

// A fragment (16x32 bf16) from row-major f32 tile; base = p + row0*ld + k0.
// Lane L<16 : row=L,    elems 0..7 -> K=0..7,  elems 8..15 -> K=16..23
// Lane L>=16: row=L-16, elems 0..7 -> K=8..15, elems 8..15 -> K=24..31
__device__ __forceinline__ v16bf load_a_f32(const float* base, int ld, int lane) {
  int row = lane & 15;
  int kb  = (lane >> 4) << 3;
  const float4* p0 = (const float4*)(base + row * ld + kb);       // 32B aligned
  const float4* p1 = (const float4*)(base + row * ld + 16 + kb);
  float4 q0 = p0[0], q1 = p0[1], q2 = p1[0], q3 = p1[1];
  v16bf a;
  a[0]  = (__bf16)q0.x; a[1]  = (__bf16)q0.y; a[2]  = (__bf16)q0.z; a[3]  = (__bf16)q0.w;
  a[4]  = (__bf16)q1.x; a[5]  = (__bf16)q1.y; a[6]  = (__bf16)q1.z; a[7]  = (__bf16)q1.w;
  a[8]  = (__bf16)q2.x; a[9]  = (__bf16)q2.y; a[10] = (__bf16)q2.z; a[11] = (__bf16)q2.w;
  a[12] = (__bf16)q3.x; a[13] = (__bf16)q3.y; a[14] = (__bf16)q3.z; a[15] = (__bf16)q3.w;
  return a;
}

__device__ __forceinline__ float sigmoidf_(float x) {
  return 1.0f / (1.0f + __expf(-x));
}

__device__ __forceinline__ unsigned wang_hash(unsigned s) {
  s = (s ^ 61u) ^ (s >> 16);
  s *= 9u;
  s ^= s >> 4;
  s *= 0x27d4eb2du;
  s ^= s >> 15;
  return s;
}

// ---------------------------------------------------------------------------
// Build WMMA B fragments for logical B[k][n] (K=128, ncols columns).
// transposed==0: B[k][n] = src[k*ncols + n]          (src is [128][ncols] f32)
// transposed==1: B[k][n] = src[n*128  + k]           (src is [ncols][128] f32)
// out[((kt*NT + nt)*32 + lane)*16 + j]
// ---------------------------------------------------------------------------
__global__ __launch_bounds__(256) void build_bfrag_kernel(
    const float* __restrict__ src, __bf16* __restrict__ out,
    int ncols, int transposed) {
  int NT = ncols >> 4;
  int total = 4 * NT * 32 * 2;  // units of 8 contiguous-k elements
  int t = blockIdx.x * blockDim.x + threadIdx.x;
  if (t >= total) return;
  int half = t & 1;
  int frag = t >> 1;            // (kt*NT + nt)*32 + lane
  int lane = frag & 31;
  int nt   = (frag >> 5) % NT;
  int kt   = (frag >> 5) / NT;
  int col  = nt * 16 + (lane & 15);
  int kb   = (lane >> 4) << 3;
  int k0   = kt * 32 + (half ? 16 + kb : kb);
  __bf16* o = out + frag * FRAG_ELEMS + half * 8;
#pragma unroll
  for (int j = 0; j < 8; ++j) {
    float v = transposed ? src[col * CCH + (k0 + j)] : src[(k0 + j) * ncols + col];
    o[j] = (__bf16)v;
  }
}

// ---------------------------------------------------------------------------
// m = bf16( h @ W )  with pre-swizzled W fragments (32 KB) staged in LDS.
// Block: 256 threads = 8 waves; wave -> one 16-row tile, all 8 column tiles.
// ---------------------------------------------------------------------------
__global__ __launch_bounds__(256) void gemm_m_kernel(
    const float* __restrict__ h, const __bf16* __restrict__ wfrag,
    __bf16* __restrict__ m, int nrows) {
  __shared__ __align__(32) __bf16 sB[4 * 8 * 32 * FRAG_ELEMS];  // 32 KB

  int tid = threadIdx.x;
  for (int i = tid; i < (4 * 8 * 32 * FRAG_ELEMS) / 8; i += blockDim.x)
    ((uint4*)sB)[i] = ((const uint4*)wfrag)[i];
  __syncthreads();

  int wave = tid >> 5, lane = tid & 31;
  int rt   = blockIdx.x * 8 + wave;
  int row0 = rt * 16;
  if (row0 >= nrows) return;  // wave-uniform

  const v16bf* sBf = (const v16bf*)sB;
  v8f acc[8] = {};
#pragma unroll
  for (int kt = 0; kt < 4; ++kt) {
    v16bf a = load_a_f32(h + (size_t)row0 * CCH + kt * 32, CCH, lane);
#pragma unroll
    for (int t = 0; t < 8; ++t) {
      v16bf b = sBf[(kt * 8 + t) * 32 + lane];  // ds_load_b128 x2
      acc[t] = wmma_bf16(a, b, acc[t]);
    }
  }

  int colL = lane & 15;
  int mrow = (lane >> 4) << 3;
#pragma unroll
  for (int t = 0; t < 8; ++t)
#pragma unroll
    for (int r = 0; r < 8; ++r)
      m[(size_t)(row0 + mrow + r) * CCH + t * 16 + colL] = (__bf16)acc[t][r];
}

// ---------------------------------------------------------------------------
// agg[dst] += m[src]  (32 lanes per edge, 4 floats per lane)
// ---------------------------------------------------------------------------
__global__ __launch_bounds__(256) void scatter_kernel(
    const __bf16* __restrict__ m, const long long* __restrict__ src,
    const long long* __restrict__ dst, float* __restrict__ agg, long long nedges) {
  long long g = (long long)blockIdx.x * blockDim.x + threadIdx.x;
  long long e = g >> 5;
  if (e >= nedges) return;
  int c = (int)(g & 31) * 4;
  long long s = src[e], d = dst[e];
  v4bf mv = *(const v4bf*)(m + (size_t)s * CCH + c);  // 8B vector gather
  float* ap = agg + (size_t)d * CCH + c;
#pragma unroll
  for (int j = 0; j < 4; ++j) atomicAdd(ap + j, (float)mv[j]);
}

// ---------------------------------------------------------------------------
// Fused GRU cell. Block = one 16-row tile: cooperatively stage bf16 A-fragments
// of agg & h into LDS once, then 8 waves each own one 16-channel tile.
// gi = agg @ WihT + b_ih ; gh = h @ WhhT + b_hh ; gate math.
// mode: 0 = plain update, 1 = relu+dropout (end of conv1), 2 = relu -> d_out
// ---------------------------------------------------------------------------
__global__ __launch_bounds__(256) void gru_kernel(
    const float* __restrict__ agg, const float* __restrict__ h_in,
    float* __restrict__ h_out, const __bf16* __restrict__ wihS,
    const __bf16* __restrict__ whhS, const float* __restrict__ b_ih,
    const float* __restrict__ b_hh, float* __restrict__ out, int mode, int nrows) {
  __shared__ __align__(32) __bf16 sAgg[4 * 32 * FRAG_ELEMS];  // 4 KB
  __shared__ __align__(32) __bf16 sH[4 * 32 * FRAG_ELEMS];    // 4 KB

  int tid  = threadIdx.x;
  int row0 = blockIdx.x * 16;
  if (row0 >= nrows) return;  // block-uniform

  {  // cooperative A-fragment staging: thread -> (kt, lane, half), 8 values
    int kt   = tid >> 6;
    int ln   = (tid >> 1) & 31;
    int half = tid & 1;
    int row  = row0 + (ln & 15);
    int kb   = (ln >> 4) << 3;
    int k0   = kt * 32 + (half ? 16 + kb : kb);
    const float4* pa = (const float4*)(agg  + (size_t)row * CCH + k0);
    const float4* ph = (const float4*)(h_in + (size_t)row * CCH + k0);
    float4 a0 = pa[0], a1 = pa[1];
    float4 b0 = ph[0], b1 = ph[1];
    v8bf va, vh;
    va[0] = (__bf16)a0.x; va[1] = (__bf16)a0.y; va[2] = (__bf16)a0.z; va[3] = (__bf16)a0.w;
    va[4] = (__bf16)a1.x; va[5] = (__bf16)a1.y; va[6] = (__bf16)a1.z; va[7] = (__bf16)a1.w;
    vh[0] = (__bf16)b0.x; vh[1] = (__bf16)b0.y; vh[2] = (__bf16)b0.z; vh[3] = (__bf16)b0.w;
    vh[4] = (__bf16)b1.x; vh[5] = (__bf16)b1.y; vh[6] = (__bf16)b1.z; vh[7] = (__bf16)b1.w;
    int off = (kt * 32 + ln) * FRAG_ELEMS + half * 8;  // 16B aligned
    *(v8bf*)(sAgg + off) = va;
    *(v8bf*)(sH + off)   = vh;
  }
  __syncthreads();

  int wave = tid >> 5, lane = tid & 31;
  int ct = wave;        // channel tile 0..7
  int c0 = ct * 16;

  const v16bf* sa = (const v16bf*)sAgg;
  const v16bf* sh = (const v16bf*)sH;
  const v16bf* bi = (const v16bf*)wihS;  // NT = 24
  const v16bf* bh = (const v16bf*)whhS;

  v8f gir = {}, giz = {}, gin = {}, ghr = {}, ghz = {}, ghn = {};
#pragma unroll
  for (int kt = 0; kt < 4; ++kt) {
    v16bf aa = sa[kt * 32 + lane];
    v16bf ah = sh[kt * 32 + lane];
    v16bf b;
    b = bi[(kt * 24 + ct) * 32 + lane];        gir = wmma_bf16(aa, b, gir);
    b = bi[(kt * 24 + 8 + ct) * 32 + lane];    giz = wmma_bf16(aa, b, giz);
    b = bi[(kt * 24 + 16 + ct) * 32 + lane];   gin = wmma_bf16(aa, b, gin);
    b = bh[(kt * 24 + ct) * 32 + lane];        ghr = wmma_bf16(ah, b, ghr);
    b = bh[(kt * 24 + 8 + ct) * 32 + lane];    ghz = wmma_bf16(ah, b, ghz);
    b = bh[(kt * 24 + 16 + ct) * 32 + lane];   ghn = wmma_bf16(ah, b, ghn);
  }

  int col = c0 + (lane & 15);
  float bir = b_ih[col], biz = b_ih[col + CCH], bin = b_ih[col + 2 * CCH];
  float bhr = b_hh[col], bhz = b_hh[col + CCH], bhn = b_hh[col + 2 * CCH];
  int mrow = (lane >> 4) << 3;

#pragma unroll
  for (int r = 0; r < 8; ++r) {
    int row = row0 + mrow + r;
    float hold = h_in[(size_t)row * CCH + col];
    float rr = sigmoidf_((gir[r] + bir) + (ghr[r] + bhr));
    float zz = sigmoidf_((giz[r] + biz) + (ghz[r] + bhz));
    float ng = tanhf((gin[r] + bin) + rr * (ghn[r] + bhn));
    float hv = (1.0f - zz) * ng + zz * hold;
    if (mode >= 1) hv = fmaxf(hv, 0.0f);
    if (mode == 1) {
      unsigned hs = wang_hash((unsigned)(row * CCH + col) ^ 0x9e3779b9u);
      hv = (hs % 10u) ? hv * (1.0f / 0.9f) : 0.0f;
    }
    if (mode == 2) out[(size_t)row * CCH + col] = hv;
    else           h_out[(size_t)row * CCH + col] = hv;
  }
}

// ---------------------------------------------------------------------------
extern "C" void kernel_launch(void* const* d_in, const int* in_sizes, int n_in,
                              void* d_out, int out_size, void* d_ws, size_t ws_size,
                              hipStream_t stream) {
  (void)n_in; (void)out_size; (void)ws_size;
  const float*     x  = (const float*)d_in[0];
  const long long* ei = (const long long*)d_in[1];  // int64 [2][E]
  const float* wL[2]   = {(const float*)d_in[2], (const float*)d_in[7]};
  const float* wihF[2] = {(const float*)d_in[3], (const float*)d_in[8]};
  const float* whhF[2] = {(const float*)d_in[4], (const float*)d_in[9]};
  const float* bih[2]  = {(const float*)d_in[5], (const float*)d_in[10]};
  const float* bhh[2]  = {(const float*)d_in[6], (const float*)d_in[11]};
  float* out = (float*)d_out;

  const int NN = in_sizes[0] / CCH;
  const long long E = in_sizes[1] / 2;
  const long long* src = ei;
  const long long* dst = ei + E;

  // ---- workspace carve-out (~64.6 MB) ----
  char* ws = (char*)d_ws;
  size_t off = 0;
  auto alloc = [&](size_t bytes) -> void* {
    void* p = ws + off;
    off += (bytes + 255) & ~(size_t)255;
    return p;
  };
  const int FR_W  = 4 * 8 * 32 * FRAG_ELEMS;   // per 128x128 layer: 16384 elems
  const int FR_G  = 4 * 24 * 32 * FRAG_ELEMS;  // per 128x384 matrix: 49152 elems
  float*  h0   = (float*)alloc((size_t)NN * CCH * 4);
  float*  h1   = (float*)alloc((size_t)NN * CCH * 4);
  float*  agg  = (float*)alloc((size_t)NN * CCH * 4);
  __bf16* mbuf = (__bf16*)alloc((size_t)NN * CCH * 2);
  __bf16* wstepS[2] = {(__bf16*)alloc((size_t)2 * FR_W * 2), (__bf16*)alloc((size_t)2 * FR_W * 2)};
  __bf16* wihS[2]   = {(__bf16*)alloc((size_t)FR_G * 2),     (__bf16*)alloc((size_t)FR_G * 2)};
  __bf16* whhS[2]   = {(__bf16*)alloc((size_t)FR_G * 2),     (__bf16*)alloc((size_t)FR_G * 2)};

  // ---- build pre-swizzled bf16 B fragments ----
  for (int cv = 0; cv < 2; ++cv) {
    for (int l = 0; l < 2; ++l)
      build_bfrag_kernel<<<(8 * 256 + 255) / 256, 256, 0, stream>>>(
          wL[cv] + (size_t)l * CCH * CCH, wstepS[cv] + (size_t)l * FR_W, CCH, 0);
    build_bfrag_kernel<<<(24 * 256 + 255) / 256, 256, 0, stream>>>(
        wihF[cv], wihS[cv], C3CH, 1);
    build_bfrag_kernel<<<(24 * 256 + 255) / 256, 256, 0, stream>>>(
        whhF[cv], whhS[cv], C3CH, 1);
  }

  hipMemcpyAsync(h0, x, (size_t)NN * CCH * 4, hipMemcpyDeviceToDevice, stream);

  float* cur = h0;
  float* nxt = h1;
  const int rowTiles   = (NN + 15) / 16;
  const int gemmBlocks = (rowTiles + 7) / 8;
  const long long sthreads = E * 32;
  const int scatBlocks = (int)((sthreads + 255) / 256);

  for (int cv = 0; cv < 2; ++cv) {
    for (int step = 0; step < 2; ++step) {
      hipMemsetAsync(agg, 0, (size_t)NN * CCH * 4, stream);
      gemm_m_kernel<<<gemmBlocks, 256, 0, stream>>>(
          cur, wstepS[cv] + (size_t)step * FR_W, mbuf, NN);
      scatter_kernel<<<scatBlocks, 256, 0, stream>>>(mbuf, src, dst, agg, E);
      int mode = (step == 1) ? ((cv == 0) ? 1 : 2) : 0;
      gru_kernel<<<rowTiles, 256, 0, stream>>>(
          agg, cur, nxt, wihS[cv], whhS[cv], bih[cv], bhh[cv], out, mode, NN);
      float* t = cur; cur = nxt; nxt = t;
    }
  }
}
